// SSSD_29996051596001
// MI455X (gfx1250) — matrix operand order repaired and checked by reading, not compiled, not run on hardware
//
#include <hip/hip_runtime.h>
#include <math.h>

// ---------------------------------------------------------------------------
// Problem constants (match reference)
// ---------------------------------------------------------------------------
constexpr int kB   = 16;
constexpr int kL   = 1024;
constexpr int kCIN = 14;
constexpr int kT   = 256;   // num_tokens
constexpr int kH   = 512;   // 2*T
constexpr int kN   = 32;    // d_state
constexpr int kE   = 128;   // embedding dim
constexpr int kD   = 6;     // depth
constexpr float kBETA0 = 1e-4f;
constexpr float kBETA1 = 2e-2f;

typedef __attribute__((ext_vector_type(16))) __bf16 v16bf;
typedef __attribute__((ext_vector_type(8)))  float  v8f;
typedef unsigned int v4u __attribute__((ext_vector_type(4)));
typedef int          v8i __attribute__((ext_vector_type(8)));
typedef int          v4i __attribute__((ext_vector_type(4)));

// ---------------------------------------------------------------------------
// Tensor Data Mover: 2D tile load Global -> LDS.
//   D# group0: count=1, lds_addr, 57-bit global addr, type=2 ("image")
//   D# group1: data_size=4B, tensor_dim0/1 (OOB zero-fill), tile_dim0 (contig
//              elems), tile_dim1 (rows), tensor_dim0_stride (row stride).
// Issued by one wave; completion via TENSORcnt.
// ---------------------------------------------------------------------------
__device__ __forceinline__ void tdm_load_2d(unsigned ldsByteOff, const float* gsrc,
                                            unsigned tileCols, unsigned tileRows,
                                            unsigned rowStrideElems,
                                            unsigned tensorD0, unsigned tensorD1)
{
    const unsigned long long ga = (unsigned long long)(const void*)gsrc;
    v4u g0;
    g0[0] = 1u;                                   // count=1 user descriptor
    g0[1] = ldsByteOff;                           // lds_addr (bytes)
    g0[2] = (unsigned)(ga & 0xFFFFFFFFull);       // global_addr[31:0]
    g0[3] = (unsigned)((ga >> 32) & 0x01FFFFFFu) | 0x80000000u; // [56:32]|type=2
    v8i g1;
    g1[0] = 0x00020000;                           // data_size = 4 bytes
    g1[1] = (int)((tensorD0 & 0xFFFFu) << 16);    // tensor_dim0[15:0] @ 63:48
    g1[2] = (int)((tensorD0 >> 16) | ((tensorD1 & 0xFFFFu) << 16));
    g1[3] = (int)((tensorD1 >> 16) | (tileCols << 16));   // tile_dim0 @127:112
    g1[4] = (int)tileRows;                        // tile_dim1 (tile_dim2 = 0)
    g1[5] = (int)rowStrideElems;                  // tensor_dim0_stride low32
    g1[6] = 0;
    g1[7] = 0;
    const v4i z4 = {0, 0, 0, 0};
    const v8i z8 = {0, 0, 0, 0, 0, 0, 0, 0};
    __builtin_amdgcn_tensor_load_to_lds(g0, g1, z4, z4, z8, 0);
}

__device__ __forceinline__ unsigned ldsOff(const void* p)
{
    return (unsigned)(unsigned long long)p;       // low 32 bits = LDS offset
}

// ---------------------------------------------------------------------------
// WMMA GEMM for conv1x1:  out[b, m, l] = epi( W[m,:] @ X[b,:,l] )
//   Block = 128 threads = 4 waves; block tile 64(M) x 32(N); each wave owns a
//   16x32 strip = 2 chained v_wmma_f32_16x16x32_bf16 per 32-wide K step.
//   INMODE 0: X = [B,K,L] activations, TDM-staged, double buffered.
//   INMODE 1: X = raw input x [B,L,CIN] (transposed read, K padded to 32).
//   EPI 0: + bias[row]                          (in_proj)
//   EPI 1: relu(acc*xscale + bias[row])         (out1; folds skip/sqrt(6))
//   EPI 2: (acc+bias)/sqrt(beta(t)) transposed  (out2 -> [B,L,CIN])
//   EPI 3: res/skip split: h=(h+v)/sqrt2, skip+=v  (op conv)
//   EPI 4: + bias2d[b,row]                      (ip conv; emb fold = ebias)
// ---------------------------------------------------------------------------
template<int KRAW, int MROWS, int INMODE, int EPI>
__global__ __launch_bounds__(128)
void gemm1x1(const float* __restrict__ W, const float* __restrict__ bias,
             const float* __restrict__ X, float* __restrict__ out,
             float* __restrict__ hbuf, float* __restrict__ skipbuf,
             const float* __restrict__ tvec, float xscale)
{
    constexpr int KT = (KRAW + 31) / 32;
    __shared__ float sA[2][64 * 32];   // W tiles  (double buffered)
    __shared__ float sX[2][32 * 32];   // X tiles  (double buffered)
    __shared__ float sD[64 * 32];      // D staging for coalesced epilogue

    const int tid  = threadIdx.x;
    const int lane = tid & 31;
    const int wv   = tid >> 5;
    const int mn   = lane & 15;
    const int g    = lane >> 4;
    const int lBase = blockIdx.x * 32;
    const int mBase = blockIdx.y * 64;
    const int b     = blockIdx.z;

    v8f acc0 = {}, acc1 = {};

    if constexpr (INMODE == 0) {
        // ---- TDM double-buffered pipeline ----
        if (tid < 32) {
            tdm_load_2d(ldsOff(sA[0]), W + (size_t)mBase * KRAW,
                        32, 64, KRAW, KRAW, MROWS);
            tdm_load_2d(ldsOff(sX[0]), X + ((size_t)b * KRAW) * kL + lBase,
                        32, 32, kL, kL, KRAW);
        }
        for (int kt = 0; kt < KT; ++kt) {
            if (tid < 32) __builtin_amdgcn_s_wait_tensorcnt(0);
            __syncthreads();
            if (tid < 32 && kt + 1 < KT) {
                const int k1 = (kt + 1) * 32;
                const int nb = (kt + 1) & 1;
                tdm_load_2d(ldsOff(sA[nb]), W + (size_t)mBase * KRAW + k1,
                            32, 64, KRAW, KRAW, MROWS);
                tdm_load_2d(ldsOff(sX[nb]),
                            X + ((size_t)b * KRAW + k1) * kL + lBase,
                            32, 32, kL, kL, KRAW);
            }
            const float* pA = sA[kt & 1];
            const float* pX = sX[kt & 1];
            v16bf af, bf0, bf1;
#pragma unroll
            for (int i = 0; i < 16; ++i) {
                const int r = i >> 1, slot = i & 1;
                const int ka = ((r & 4) ? 16 : 0) + (g << 3) + ((r & 3) << 1) + slot;
                af[i]  = (__bf16)pA[(wv * 16 + mn) * 32 + ka];
                const int kb = ((i & 8) ? 16 : 0) + (g << 3) + (i & 7);
                bf0[i] = (__bf16)pX[kb * 32 + mn];
                bf1[i] = (__bf16)pX[kb * 32 + 16 + mn];
            }
            acc0 = __builtin_amdgcn_wmma_f32_16x16x32_bf16(
                false, af, false, bf0, (short)0, acc0, false, false);
            acc1 = __builtin_amdgcn_wmma_f32_16x16x32_bf16(
                false, af, false, bf1, (short)0, acc1, false, false);
        }
    } else {
        // ---- INMODE 1: transposed read of x[B,L,CIN], K=14 -> one step ----
        {
            const int row = tid >> 1, half = tid & 1;
            const int gr  = mBase + row;
            for (int j = 0; j < 16; ++j) {
                const int kk = half * 16 + j;
                sA[0][row * 32 + half * 16 + j] =
                    (gr < MROWS && kk < KRAW) ? W[(size_t)gr * KRAW + kk] : 0.f;
            }
            const int k = tid & 31, cg = tid >> 5;
            for (int j = 0; j < 8; ++j) {
                const int c = cg * 8 + j;
                sX[0][k * 32 + c] =
                    (k < KRAW) ? X[((size_t)b * kL + lBase + c) * KRAW + k] : 0.f;
            }
        }
        __syncthreads();
        v16bf af, bf0, bf1;
#pragma unroll
        for (int i = 0; i < 16; ++i) {
            const int r = i >> 1, slot = i & 1;
            const int ka = ((r & 4) ? 16 : 0) + (g << 3) + ((r & 3) << 1) + slot;
            af[i]  = (__bf16)sA[0][(wv * 16 + mn) * 32 + ka];
            const int kb = ((i & 8) ? 16 : 0) + (g << 3) + (i & 7);
            bf0[i] = (__bf16)sX[0][kb * 32 + mn];
            bf1[i] = (__bf16)sX[0][kb * 32 + 16 + mn];
        }
        acc0 = __builtin_amdgcn_wmma_f32_16x16x32_bf16(
            false, af, false, bf0, (short)0, acc0, false, false);
        acc1 = __builtin_amdgcn_wmma_f32_16x16x32_bf16(
            false, af, false, bf1, (short)0, acc1, false, false);
    }

    // ---- transpose D through LDS for coalesced b128 epilogue stores ----
    __syncthreads();
#pragma unroll
    for (int r = 0; r < 8; ++r) {
        const int rl = (wv * 16 + (g << 3) + r) * 32;
        sD[rl + mn]      = acc0[r];
        sD[rl + 16 + mn] = acc1[r];
    }
    __syncthreads();

    const int row  = tid >> 1;
    const int half = tid & 1;
    const int grow = mBase + row;
    const int col0 = lBase + half * 16;
    const float* dv = sD + row * 32 + half * 16;

    if constexpr (EPI == 2) {
        if (grow < kCIN) {
            const float rb = rsqrtf(kBETA0 + tvec[b] * (kBETA1 - kBETA0));
            const float b0 = bias[grow];
            for (int j = 0; j < 16; ++j)
                out[((size_t)b * kL + col0 + j) * kCIN + grow] = (dv[j] + b0) * rb;
        }
    } else if constexpr (EPI == 3) {
        const float b0 = bias[grow];
        if (grow < kT) {
            float4* hp = (float4*)(hbuf + ((size_t)b * kT + grow) * kL + col0);
#pragma unroll
            for (int q = 0; q < 4; ++q) {
                float4 hv = hp[q];
                hv.x = (hv.x + dv[q * 4 + 0] + b0) * 0.70710678118654752f;
                hv.y = (hv.y + dv[q * 4 + 1] + b0) * 0.70710678118654752f;
                hv.z = (hv.z + dv[q * 4 + 2] + b0) * 0.70710678118654752f;
                hv.w = (hv.w + dv[q * 4 + 3] + b0) * 0.70710678118654752f;
                hp[q] = hv;
            }
        } else {
            float4* sp = (float4*)(skipbuf + ((size_t)b * kT + (grow - kT)) * kL + col0);
#pragma unroll
            for (int q = 0; q < 4; ++q) {
                float4 sv = sp[q];
                sv.x += dv[q * 4 + 0] + b0; sv.y += dv[q * 4 + 1] + b0;
                sv.z += dv[q * 4 + 2] + b0; sv.w += dv[q * 4 + 3] + b0;
                sp[q] = sv;
            }
        }
    } else {
        const float b0 = (EPI == 4) ? bias[(size_t)b * MROWS + grow] : bias[grow];
        float4* op = (float4*)(out + ((size_t)b * MROWS + grow) * kL + col0);
#pragma unroll
        for (int q = 0; q < 4; ++q) {
            float4 v;
            if constexpr (EPI == 1) {
                v.x = fmaxf(dv[q * 4 + 0] * xscale + b0, 0.f);
                v.y = fmaxf(dv[q * 4 + 1] * xscale + b0, 0.f);
                v.z = fmaxf(dv[q * 4 + 2] * xscale + b0, 0.f);
                v.w = fmaxf(dv[q * 4 + 3] * xscale + b0, 0.f);
            } else {
                v.x = dv[q * 4 + 0] + b0; v.y = dv[q * 4 + 1] + b0;
                v.z = dv[q * 4 + 2] + b0; v.w = dv[q * 4 + 3] + b0;
            }
            op[q] = v;
        }
    }
}

// ---------------------------------------------------------------------------
// S4D as a 32-state complex linear recurrence (exact rewrite of the FFT conv):
//   s_n[l] = lam_n*s_n[l-1] + u[l];  y[l] = Re(sum_n 2*Ct_n*s_n[l]) + D*u[l]
// 4 lanes per (b,h) channel, 8 complex states per lane, shfl_xor combine.
// ---------------------------------------------------------------------------
__global__ __launch_bounds__(256)
void s4_scan(const float* __restrict__ u, float* __restrict__ y,
             const float* __restrict__ P,   // [H][N][4] = {lre, lim, 2Cre, 2Cim}
             const float* __restrict__ Dv)  // [H]
{
    const int t    = blockIdx.x * 256 + threadIdx.x;
    const int sub  = t & 3;
    const int pair = t >> 2;              // b*H + h
    const int h    = pair & (kH - 1);
    const float* up = u + (size_t)pair * kL;
    float*       yp = y + (size_t)pair * kL;

    float lre[8], lim[8], cre[8], cim[8], sr[8], si[8];
    const float4* pp = (const float4*)(P + ((size_t)h * kN + sub * 8) * 4);
#pragma unroll
    for (int n = 0; n < 8; ++n) {
        const float4 q = pp[n];
        lre[n] = q.x; lim[n] = q.y; cre[n] = q.z; cim[n] = q.w;
        sr[n] = 0.f;  si[n] = 0.f;
    }
    const float Dh = Dv[h];

    for (int l = 0; l < kL; ++l) {
        const float uv = up[l];
        float acc = 0.f;
#pragma unroll
        for (int n = 0; n < 8; ++n) {
            const float nr = fmaf(lre[n], sr[n], fmaf(-lim[n], si[n], uv));
            const float ni = fmaf(lre[n], si[n], lim[n] * sr[n]);
            sr[n] = nr; si[n] = ni;
            acc = fmaf(cre[n], nr, fmaf(-cim[n], ni, acc));
        }
        acc += __shfl_xor(acc, 1);
        acc += __shfl_xor(acc, 2);
        if (sub == 0) yp[l] = acc + Dh * uv;
    }
}

// ---------------------------------------------------------------------------
// Precompute per-(depth,slot,h,n): lam = exp(dt*A), Ct = 2*(Cre+iCim)*(lam-1)/A
// ---------------------------------------------------------------------------
__global__ void s4_pre(const float* __restrict__ log_dt,
                       const float* __restrict__ logA_re,
                       const float* __restrict__ A_im,
                       const float* __restrict__ C_re,
                       const float* __restrict__ C_im,
                       float* __restrict__ P)
{
    const int idx = blockIdx.x * 256 + threadIdx.x;      // D*2*H*N
    if (idx >= kD * 2 * kH * kN) return;
    const int h  = (idx >> 5) & (kH - 1);
    const int ds = idx >> 14;                            // H*N = 2^14
    const float dt  = expf(log_dt[ds * kH + h]);
    const float Are = -expf(logA_re[idx]);
    const float Aim = A_im[idx];
    const float er  = expf(Are * dt);
    const float lre = er * cosf(Aim * dt);
    const float lim = er * sinf(Aim * dt);
    const float d2  = Are * Are + Aim * Aim;
    const float fr  = ((lre - 1.f) * Are + lim * Aim) / d2;
    const float fi  = (lim * Are - (lre - 1.f) * Aim) / d2;
    const float cr  = C_re[idx] * fr - C_im[idx] * fi;
    const float ci  = C_re[idx] * fi + C_im[idx] * fr;
    float4 q = {lre, lim, 2.f * cr, 2.f * ci};
    *(float4*)(P + (size_t)idx * 4) = q;
}

// ---------------------------------------------------------------------------
// Channel LayerNorm over H=512 per (b,l) column; one wave per column.
// GATE=true fuses sigmoid(gate)*tanh(filt) (channel o pairs with o+256 in lane)
// ---------------------------------------------------------------------------
template<bool GATE>
__global__ __launch_bounds__(256)
void ln_chan(const float* __restrict__ in, float* __restrict__ out,
             const float* __restrict__ gv, const float* __restrict__ bv)
{
    const int wv   = threadIdx.x >> 5;
    const int lane = threadIdx.x & 31;
    const int colIdx = blockIdx.x * 8 + wv;
    const int b = colIdx >> 10;
    const int l = colIdx & (kL - 1);
    const float* base = in + (size_t)b * kH * kL + l;

    float v[16];
    float s = 0.f, ss = 0.f;
#pragma unroll
    for (int i = 0; i < 16; ++i) {
        v[i] = base[(size_t)(lane + 32 * i) * kL];
        s += v[i]; ss += v[i] * v[i];
    }
#pragma unroll
    for (int off = 16; off >= 1; off >>= 1) {
        s  += __shfl_xor(s, off);
        ss += __shfl_xor(ss, off);
    }
    const float mean = s * (1.f / kH);
    const float var  = ss * (1.f / kH) - mean * mean;
    const float rstd = rsqrtf(var + 1e-5f);

    if constexpr (!GATE) {
#pragma unroll
        for (int i = 0; i < 16; ++i) {
            const int hc = lane + 32 * i;
            out[(size_t)b * kH * kL + (size_t)hc * kL + l] =
                (v[i] - mean) * rstd * gv[hc] + bv[hc];
        }
    } else {
        float nv[16];
#pragma unroll
        for (int i = 0; i < 16; ++i) {
            const int hc = lane + 32 * i;
            nv[i] = (v[i] - mean) * rstd * gv[hc] + bv[hc];
        }
#pragma unroll
        for (int i = 0; i < 8; ++i) {
            const int o = lane + 32 * i;          // gate channel; filt = o+256
            const float gate = 1.f / (1.f + expf(-nv[i]));
            out[((size_t)b * kT + o) * kL + l] = gate * tanhf(nv[i + 8]);
        }
    }
}

// ---------------------------------------------------------------------------
// Gaussian Fourier projection + 2-layer SiLU MLP -> emb[B,E]
// ---------------------------------------------------------------------------
__global__ __launch_bounds__(128)
void embed_k(const float* __restrict__ t, const float* __restrict__ gfp,
             const float* __restrict__ w1, const float* __restrict__ b1,
             const float* __restrict__ w2, const float* __restrict__ b2,
             float* __restrict__ emb)
{
    __shared__ float v0[kE], v1[kE];
    const int b = blockIdx.x, e = threadIdx.x;
    const float tv = t[b];
    if (e < kE / 2) v0[e] = sinf(tv * gfp[e] * 6.283185307179586f);
    else            v0[e] = cosf(tv * gfp[e - kE / 2] * 6.283185307179586f);
    __syncthreads();
    float a = b1[e];
    for (int j = 0; j < kE; ++j) a += w1[e * kE + j] * v0[j];
    a = a / (1.f + expf(-a));
    v1[e] = a;
    __syncthreads();
    float c = b2[e];
    for (int j = 0; j < kE; ++j) c += w2[e * kE + j] * v1[j];
    emb[b * kE + e] = c / (1.f + expf(-c));
}

// embp[d][b][o] = emb[b] . dp_w[d][o] + dp_b[d][o]
__global__ void embp_k(const float* __restrict__ emb, const float* __restrict__ dp_w,
                       const float* __restrict__ dp_b, float* __restrict__ embp)
{
    const int idx = blockIdx.x * 256 + threadIdx.x;      // D*B*T
    if (idx >= kD * kB * kT) return;
    const int d = idx / (kB * kT);
    const int rem = idx - d * (kB * kT);
    const int b = rem / kT;
    const int o = rem - b * kT;
    float a = dp_b[d * kT + o];
    const float* wrow = dp_w + ((size_t)d * kT + o) * kE;
    const float* ev   = emb + b * kE;
    for (int j = 0; j < kE; ++j) a += wrow[j] * ev[j];
    embp[idx] = a;
}

// ebias[d][b][m] = ip_w[d][m,:] . embp[d][b,:] + ip_b[d][m]
// (folds the broadcast "h + emb" add through the ip conv into a 2D bias)
__global__ void ebias_k(const float* __restrict__ embp, const float* __restrict__ ip_w,
                        const float* __restrict__ ip_b, float* __restrict__ ebias)
{
    const int idx = blockIdx.x * 256 + threadIdx.x;      // D*B*H
    if (idx >= kD * kB * kH) return;
    const int d = idx / (kB * kH);
    const int rem = idx - d * (kB * kH);
    const int b = rem / kH;
    const int m = rem - b * kH;
    const float* wrow = ip_w + ((size_t)d * kH + m) * kT;
    const float* ev   = embp + ((size_t)d * kB + b) * kT;
    float a = ip_b[d * kH + m];
    for (int j = 0; j < kT; ++j) a += wrow[j] * ev[j];
    ebias[idx] = a;
}

// ---------------------------------------------------------------------------
extern "C" void kernel_launch(void* const* d_in, const int* in_sizes, int n_in,
                              void* d_out, int out_size, void* d_ws, size_t ws_size,
                              hipStream_t stream)
{
    const float* x       = (const float*)d_in[0];
    const float* t       = (const float*)d_in[1];
    const float* gfp_W   = (const float*)d_in[2];
    const float* emb_w1  = (const float*)d_in[3];
    const float* emb_b1  = (const float*)d_in[4];
    const float* emb_w2  = (const float*)d_in[5];
    const float* emb_b2  = (const float*)d_in[6];
    const float* in_w    = (const float*)d_in[7];
    const float* in_b    = (const float*)d_in[8];
    const float* out1_w  = (const float*)d_in[9];
    const float* out1_b  = (const float*)d_in[10];
    const float* out2_w  = (const float*)d_in[11];
    const float* out2_b  = (const float*)d_in[12];
    const float* dp_w    = (const float*)d_in[13];
    const float* dp_b    = (const float*)d_in[14];
    const float* ip_w    = (const float*)d_in[15];
    const float* ip_b    = (const float*)d_in[16];
    const float* op_w    = (const float*)d_in[17];
    const float* op_b    = (const float*)d_in[18];
    const float* s4_log_dt  = (const float*)d_in[19];
    const float* s4_logA_re = (const float*)d_in[20];
    const float* s4_A_im    = (const float*)d_in[21];
    const float* s4_C_re    = (const float*)d_in[22];
    const float* s4_C_im    = (const float*)d_in[23];
    const float* s4_D       = (const float*)d_in[24];
    const float* ln_g       = (const float*)d_in[25];
    const float* ln_b       = (const float*)d_in[26];
    float* dout = (float*)d_out;

    // workspace layout (floats)
    float* ws    = (float*)d_ws;
    float* emb   = ws;                                    // [B,E]
    float* embp  = emb   + kB * kE;                       // [D,B,T]
    float* ebias = embp  + kD * kB * kT;                  // [D,B,H]
    float* s4p   = ebias + kD * kB * kH;                  // [D,2,H,N,4]
    float* hb    = s4p   + (size_t)kD * 2 * kH * kN * 4;  // [B,T,L]
    float* skip  = hb    + (size_t)kB * kT * kL;          // [B,T,L]
    float* y0    = skip  + (size_t)kB * kT * kL;          // [B,H,L]
    float* y1    = y0    + (size_t)kB * kH * kL;          // [B,H,L]
    float* gbuf  = y0;   // gated [B,T,L]  (y0 dead after 2nd scan)
    float* z0    = y1;   // out1 result    (y1 dead after gating)

    (void)hipMemsetAsync(skip, 0, (size_t)kB * kT * kL * sizeof(float), stream);

    embed_k<<<kB, 128, 0, stream>>>(t, gfp_W, emb_w1, emb_b1, emb_w2, emb_b2, emb);
    embp_k<<<(kD * kB * kT + 255) / 256, 256, 0, stream>>>(emb, dp_w, dp_b, embp);
    ebias_k<<<(kD * kB * kH + 255) / 256, 256, 0, stream>>>(embp, ip_w, ip_b, ebias);
    s4_pre<<<(kD * 2 * kH * kN + 255) / 256, 256, 0, stream>>>(
        s4_log_dt, s4_logA_re, s4_A_im, s4_C_re, s4_C_im, s4p);

    const dim3 gT(kL / 32, kT / 64, kB);   // M=256 grids
    const dim3 gH(kL / 32, kH / 64, kB);   // M=512 grids
    const dim3 gC(kL / 32, 1, kB);         // M=14  grid

    // h = in_w @ x^T + in_b
    gemm1x1<kCIN, kT, 1, 0><<<gT, 128, 0, stream>>>(
        in_w, in_b, x, hb, nullptr, nullptr, nullptr, 1.f);

    const int scanBlocks = (kB * kH * 4) / 256;   // 128
    const int lnBlocks   = (kB * kL) / 8;         // 2048

    for (int d = 0; d < kD; ++d) {
        // y0 = ip_w[d] @ h + ebias[d]   (emb broadcast folded into 2D bias)
        gemm1x1<kT, kH, 0, 4><<<gH, 128, 0, stream>>>(
            ip_w + (size_t)d * kH * kT, ebias + (size_t)d * kB * kH, hb,
            y0, nullptr, nullptr, nullptr, 1.f);
        // y1 = S4D(y0) ; y0 = LN(y1)
        s4_scan<<<scanBlocks, 256, 0, stream>>>(
            y0, y1, s4p + ((size_t)(d * 2 + 0)) * kH * kN * 4,
            s4_D + (d * 2 + 0) * kH);
        ln_chan<false><<<lnBlocks, 256, 0, stream>>>(
            y1, y0, ln_g + (d * 2 + 0) * kH, ln_b + (d * 2 + 0) * kH);
        // y1 = S4D(y0) ; gbuf = sigmoid/tanh gate of LN(y1)
        s4_scan<<<scanBlocks, 256, 0, stream>>>(
            y0, y1, s4p + ((size_t)(d * 2 + 1)) * kH * kN * 4,
            s4_D + (d * 2 + 1) * kH);
        ln_chan<true><<<lnBlocks, 256, 0, stream>>>(
            y1, gbuf, ln_g + (d * 2 + 1) * kH, ln_b + (d * 2 + 1) * kH);
        // op conv with fused residual/skip epilogue
        gemm1x1<kT, kH, 0, 3><<<gH, 128, 0, stream>>>(
            op_w + (size_t)d * kH * kT, op_b + d * kH, gbuf,
            nullptr, hb, skip, nullptr, 1.f);
    }

    // z0 = relu(out1_w @ skip * (1/sqrt(6)) + out1_b)
    gemm1x1<kT, kT, 0, 1><<<gT, 128, 0, stream>>>(
        out1_w, out1_b, skip, z0, nullptr, nullptr, nullptr,
        0.40824829046386301636f);
    // out = (out2_w @ z0 + out2_b) / sqrt(beta(t)), stored as [B,L,CIN]
    gemm1x1<kT, kCIN, 0, 2><<<gC, 128, 0, stream>>>(
        out2_w, out2_b, z0, dout, nullptr, nullptr, t, 1.f);

    (void)in_sizes; (void)n_in; (void)out_size; (void)ws_size;
}
